// PortableLoftQLayer_68539088109776
// MI455X (gfx1250) — compile-verified, matrix operand order checked by reading, not run on hardware
//
#include <hip/hip_runtime.h>

typedef __attribute__((ext_vector_type(16))) __bf16        v16bf;
typedef __attribute__((ext_vector_type(8)))  float         v8f;
typedef __attribute__((ext_vector_type(4)))  unsigned int  v4u;
typedef __attribute__((ext_vector_type(2)))  unsigned int  v2u;

namespace {
constexpr int TOK  = 8192;
constexpr int INF  = 4096;   // in_features (K)
constexpr int OUTF = 4096;   // out_features (N)
constexpr int RK   = 64;     // LoRA rank (K-extension)
constexpr int LST  = 72;     // LDS row stride in u16 (64 + pad: spreads 64 banks, keeps 16B align)
}

__constant__ float NF4TAB[16] = {
    -1.0f, -0.6961928009986877f, -0.5250730514526367f, -0.39491748809814453f,
    -0.28444138169288635f, -0.18477343022823334f, -0.09105003625154495f, 0.0f,
    0.07958029955625534f, 0.16093020141124725f, 0.2461123913526535f,
    0.33791524171829224f, 0.44070982933044434f, 0.5626170039176941f,
    0.7229568362236023f, 1.0f};

union ABf { v4u q[2]; v16bf v; };

static __device__ __forceinline__ unsigned int f2bf(float f) {
  unsigned int u = __float_as_uint(f);
  u += 0x7FFFu + ((u >> 16) & 1u);          // round-to-nearest-even to bf16
  return u >> 16;
}
static __device__ __forceinline__ unsigned int pack2bf(float a, float b) {
  return f2bf(a) | (f2bf(b) << 16);
}
// NF4 codebook lookup: table value lives in lanes 0..15, one ds_bpermute per code.
static __device__ __forceinline__ float nf4_deq(int code, int tblbits) {
  return __int_as_float(__builtin_amdgcn_ds_bpermute((code & 15) << 2, tblbits));
}

// ---------------------------------------------------------------------------
// Kernel 1: T[8192,64] = x[8192,4096] @ lora_A^T[4096,64]   (fp32 out to ws)
// 128 tokens per block, full R=64 as N. 8 waves: wave w owns rows w*16..w*16+15.
// ---------------------------------------------------------------------------
__global__ __launch_bounds__(256) void lora_xA_kernel(
    const float* __restrict__ x, const float* __restrict__ loraA,
    float* __restrict__ T)
{
  __shared__ unsigned short Xs[128 * LST];
  __shared__ unsigned short As[64 * LST];

  const int tid  = threadIdx.x;
  const int lane = tid & 31;
  const int wave = tid >> 5;
  const int m0   = blockIdx.x * 128;

  const int lk   = tid & 15;     // which float4 chunk along K
  const int lrow = tid >> 4;     // base row for staging

  const int r     = lane & 15;        // row/col within fragment
  const int khalf = (lane >> 4) * 8;  // upper lanes hold K+8 per bf16 A layout

  v8f acc[4] = {};

  for (int kt = 0; kt < INF / 64; ++kt) {
    const int k0 = kt * 64;
    if (kt + 1 < INF / 64) {
      __builtin_prefetch(x + (size_t)m0 * INF + k0 + 64, 0, 3);
      __builtin_prefetch(loraA + (size_t)0 * INF + k0 + 64, 0, 3);
    }
#pragma unroll
    for (int p = 0; p < 8; ++p) {        // stage x tile 128x64 -> bf16
      const int row = lrow + p * 16;
      float4 xv = *(const float4*)(x + (size_t)(m0 + row) * INF + k0 + lk * 4);
      *(v2u*)&Xs[row * LST + lk * 4] = (v2u){pack2bf(xv.x, xv.y), pack2bf(xv.z, xv.w)};
    }
#pragma unroll
    for (int p = 0; p < 4; ++p) {        // stage lora_A tile 64x64 -> bf16
      const int row = lrow + p * 16;
      float4 av = *(const float4*)(loraA + (size_t)row * INF + k0 + lk * 4);
      *(v2u*)&As[row * LST + lk * 4] = (v2u){pack2bf(av.x, av.y), pack2bf(av.z, av.w)};
    }
    __syncthreads();

#pragma unroll
    for (int ks = 0; ks < 2; ++ks) {
      ABf a;
      const unsigned short* ab = &Xs[(wave * 16 + r) * LST + ks * 32 + khalf];
      a.q[0] = *(const v4u*)(ab);
      a.q[1] = *(const v4u*)(ab + 16);
#pragma unroll
      for (int ni = 0; ni < 4; ++ni) {
        ABf b;
        const unsigned short* bb = &As[(ni * 16 + r) * LST + ks * 32 + khalf];
        b.q[0] = *(const v4u*)(bb);
        b.q[1] = *(const v4u*)(bb + 16);
        acc[ni] = __builtin_amdgcn_wmma_f32_16x16x32_bf16(
            false, a.v, false, b.v, (short)0, acc[ni], false, false);
      }
    }
    __syncthreads();
  }

  const int hv = (lane >> 4) * 8;   // C/D layout: upper lanes are rows M+8
#pragma unroll
  for (int ni = 0; ni < 4; ++ni) {
    const int col = ni * 16 + r;
#pragma unroll
    for (int v = 0; v < 8; ++v) {
      T[(size_t)(m0 + wave * 16 + hv + v) * RK + col] = acc[ni][v];
    }
  }
}

// ---------------------------------------------------------------------------
// Kernel 2: out = x @ dequant_nf4(qcodes,absmax)^T  +  T @ lora_B^T  + bias
// Block tile 256(M) x 128(N); 8 waves as 4(M)x2(N); wave tile 64x64 = 4x4 WMMA
// fragments (16 WMMAs per 16 ds_load_b128, each fragment reused 4x).
// Main K loop of 64 tiles over W, then one extra K-tile (R=64) over (T,lora_B)
// folded into the same accumulators.
// ---------------------------------------------------------------------------
__global__ __launch_bounds__(256) void nf4_lora_gemm_kernel(
    const float* __restrict__ x, const int* __restrict__ qc,
    const float* __restrict__ absmax, const float* __restrict__ loraB,
    const float* __restrict__ bias, const float* __restrict__ T,
    float* __restrict__ out)
{
  __shared__ unsigned short Xs[256 * LST];   // 36 KB
  __shared__ unsigned short Ws[128 * LST];   // 18 KB

  const int tid  = threadIdx.x;
  const int lane = tid & 31;
  const int wave = tid >> 5;
  const int wm   = wave & 3;       // M sub-tile * 64
  const int wn   = wave >> 2;      // N sub-tile * 64
  const int m0   = blockIdx.y * 256;
  const int n0   = blockIdx.x * 128;

  const int tblbits = __float_as_int(NF4TAB[lane & 15]);  // codebook in lanes 0..15

  const int lk   = tid & 15;
  const int lrow = tid >> 4;
  const int r     = lane & 15;
  const int khalf = (lane >> 4) * 8;

  v8f acc[4][4] = {};

  for (int kt = 0; kt <= INF / 64; ++kt) {
    if (kt < INF / 64) {
      const int k0 = kt * 64;
      if (kt + 1 < INF / 64) {
        __builtin_prefetch(x + (size_t)m0 * INF + k0 + 64, 0, 3);
        __builtin_prefetch(qc + (size_t)n0 * INF + k0 + 64, 0, 3);
      }
#pragma unroll
      for (int p = 0; p < 16; ++p) {        // x tile 256x64: fp32 -> bf16
        const int row = lrow + p * 16;
        float4 xv = *(const float4*)(x + (size_t)(m0 + row) * INF + k0 + lk * 4);
        *(v2u*)&Xs[row * LST + lk * 4] = (v2u){pack2bf(xv.x, xv.y), pack2bf(xv.z, xv.w)};
      }
#pragma unroll
      for (int p = 0; p < 8; ++p) {         // W tile 128x64: NF4 dequant -> bf16
        const int row = lrow + p * 16;
        int4 qv = *(const int4*)(qc + (size_t)(n0 + row) * INF + k0 + lk * 4);
        // block index = (o*IN_F + k)/64 = o*64 + kt  (KTILE == BLOCK == 64)
        const float am = absmax[(size_t)(n0 + row) * (INF / 64) + kt];
        const float w0 = nf4_deq(qv.x, tblbits) * am;
        const float w1 = nf4_deq(qv.y, tblbits) * am;
        const float w2 = nf4_deq(qv.z, tblbits) * am;
        const float w3 = nf4_deq(qv.w, tblbits) * am;
        *(v2u*)&Ws[row * LST + lk * 4] = (v2u){pack2bf(w0, w1), pack2bf(w2, w3)};
      }
    } else {
      // K-extension: A <- T[tok, 64], B <- lora_B[out, 64]
#pragma unroll
      for (int p = 0; p < 16; ++p) {
        const int row = lrow + p * 16;
        float4 tv = *(const float4*)(T + (size_t)(m0 + row) * RK + lk * 4);
        *(v2u*)&Xs[row * LST + lk * 4] = (v2u){pack2bf(tv.x, tv.y), pack2bf(tv.z, tv.w)};
      }
#pragma unroll
      for (int p = 0; p < 8; ++p) {
        const int row = lrow + p * 16;
        float4 bv = *(const float4*)(loraB + (size_t)(n0 + row) * RK + lk * 4);
        *(v2u*)&Ws[row * LST + lk * 4] = (v2u){pack2bf(bv.x, bv.y), pack2bf(bv.z, bv.w)};
      }
    }
    __syncthreads();

#pragma unroll
    for (int ks = 0; ks < 2; ++ks) {
      ABf a[4], b[4];
#pragma unroll
      for (int mi = 0; mi < 4; ++mi) {
        const unsigned short* ab = &Xs[(wm * 64 + mi * 16 + r) * LST + ks * 32 + khalf];
        a[mi].q[0] = *(const v4u*)(ab);
        a[mi].q[1] = *(const v4u*)(ab + 16);
      }
#pragma unroll
      for (int ni = 0; ni < 4; ++ni) {
        const unsigned short* bb = &Ws[(wn * 64 + ni * 16 + r) * LST + ks * 32 + khalf];
        b[ni].q[0] = *(const v4u*)(bb);
        b[ni].q[1] = *(const v4u*)(bb + 16);
      }
#pragma unroll
      for (int mi = 0; mi < 4; ++mi)
#pragma unroll
        for (int ni = 0; ni < 4; ++ni)
          acc[mi][ni] = __builtin_amdgcn_wmma_f32_16x16x32_bf16(
              false, a[mi].v, false, b[ni].v, (short)0, acc[mi][ni], false, false);
    }
    __syncthreads();
  }

  // Epilogue: + bias, store fp32
  const int hv = (lane >> 4) * 8;
#pragma unroll
  for (int ni = 0; ni < 4; ++ni) {
    const int col = n0 + wn * 64 + ni * 16 + r;
    const float bv = bias[col];
#pragma unroll
    for (int mi = 0; mi < 4; ++mi) {
      const int rowb = m0 + wm * 64 + mi * 16 + hv;
#pragma unroll
      for (int v = 0; v < 8; ++v) {
        out[(size_t)(rowb + v) * OUTF + col] = acc[mi][ni][v] + bv;
      }
    }
  }
}

// ---------------------------------------------------------------------------
extern "C" void kernel_launch(void* const* d_in, const int* in_sizes, int n_in,
                              void* d_out, int out_size, void* d_ws, size_t ws_size,
                              hipStream_t stream) {
  (void)in_sizes; (void)n_in; (void)out_size; (void)ws_size;
  const float* x      = (const float*)d_in[0];
  const int*   qcodes = (const int*)d_in[1];
  const float* absmax = (const float*)d_in[2];
  const float* loraA  = (const float*)d_in[3];
  const float* loraB  = (const float*)d_in[4];
  const float* bias   = (const float*)d_in[5];
  float*       out    = (float*)d_out;
  float*       T      = (float*)d_ws;   // 8192 x 64 fp32 = 2 MB scratch

  lora_xA_kernel<<<dim3(TOK / 128), dim3(256), 0, stream>>>(x, loraA, T);
  nf4_lora_gemm_kernel<<<dim3(OUTF / 128, TOK / 256), dim3(256), 0, stream>>>(
      x, qcodes, absmax, loraB, bias, T, out);
}
